// FlashAttention_81518479278193
// MI455X (gfx1250) — matrix operand order repaired and checked by reading, not compiled, not run on hardware
//
#include <hip/hip_runtime.h>

// ---------------------------------------------------------------------------
// CDNA5 (gfx1250) attention block, bf16 WMMA end-to-end:
//   cvt(x,Wqkv,Wout)->bf16 ; qkv GEMM ; RoPE->bf16 Q/K/V ; flash attention
//   (TDM K-tiles, double-buffered LDS) ; out projection GEMM.
// GEMM W-tiles: global_load_async_to_lds_b128 + s_wait_asynccnt, ping-pong.
// ---------------------------------------------------------------------------

typedef __attribute__((ext_vector_type(16))) __bf16 v16bf;
typedef __attribute__((ext_vector_type(8)))  float  v8f;
typedef __attribute__((ext_vector_type(4)))  unsigned v4u;
typedef __attribute__((ext_vector_type(8)))  int      v8i;
typedef __attribute__((ext_vector_type(4)))  int      v4i;

#define SEQ   2048
#define EMB   1024
#define NHEAD 16
#define HDIM  64

// 16-bit A 16x32: lane<16 K={0..7,16..23}, lane>=16 +8 -> two 16B runs
// 16-bit B 32x16: same K order as A -> two 16B runs
__device__ __forceinline__ v16bf frag2x16(const __bf16* p0, const __bf16* p1) {
  v16bf r;
  *(uint4*)&r              = *(const uint4*)p0;
  *((uint4*)((char*)&r + 16)) = *(const uint4*)p1;
  return r;
}
__device__ __forceinline__ v16bf frag_a(const __bf16* row, int half) {
  return frag2x16(row + 8 * half, row + 16 + 8 * half);
}
__device__ __forceinline__ v16bf frag_b(const __bf16* row, int half) {
  return frag2x16(row + 8 * half, row + 16 + 8 * half);
}

// ---------------------------------------------------------------------------
// fp32 -> bf16 bulk convert (4 elems/thread)
// ---------------------------------------------------------------------------
__global__ __launch_bounds__(256) void cvt4(const float* __restrict__ src,
                                            __bf16* __restrict__ dst, int n4) {
  int i = blockIdx.x * blockDim.x + threadIdx.x;
  if (i >= n4) return;
  float4 f = ((const float4*)src)[i];
  dst[4 * i + 0] = (__bf16)f.x;
  dst[4 * i + 1] = (__bf16)f.y;
  dst[4 * i + 2] = (__bf16)f.z;
  dst[4 * i + 3] = (__bf16)f.w;
}

// ---------------------------------------------------------------------------
// GEMM: C[m,n] = sum_k A[m,k]*W[n,k], A/W bf16 row-major, C f32.
// 256 thr = 8 waves, 128x64 tile. W tile async-copied to LDS, double-buffered.
// ---------------------------------------------------------------------------
#define BS_LD 40  // bf16 elems per LDS row (80B: 16B aligned, bank-spread)

__global__ __launch_bounds__(256) void gemm_bf16(const __bf16* __restrict__ A,
                                                 const __bf16* __restrict__ W,
                                                 float* __restrict__ C,
                                                 int M, int N, int K) {
  __shared__ __align__(16) __bf16 Bs[2][64 * BS_LD];

  const int tid  = threadIdx.x;
  const int lane = tid & 31;
  const int wave = tid >> 5;
  const int half = lane >> 4;
  const int col  = lane & 15;
  const int nb   = blockIdx.x * 64;
  const int mb   = blockIdx.y * 128;
  const int mw   = mb + wave * 16;

  // async copy of one 64x32 W tile: each thread moves 16B
  const int wn = tid >> 2, wc = tid & 3;
  auto issue = [&](int buf, int k0) {
    unsigned long long g =
        (unsigned long long)(uintptr_t)(W + (size_t)(nb + wn) * K + k0 + wc * 8);
    unsigned lds = (unsigned)(uintptr_t)&Bs[buf][wn * BS_LD + wc * 8];
    asm volatile("global_load_async_to_lds_b128 %0, %1, off"
                 :: "v"(lds), "v"(g) : "memory");
  };

  v8f acc[4] = {v8f{}, v8f{}, v8f{}, v8f{}};

  issue(0, 0);
  for (int k0 = 0; k0 < K; k0 += 32) {
    const int buf = (k0 >> 5) & 1;
    asm volatile("s_wait_asynccnt 0x0" ::: "memory");
    __syncthreads();
    if (k0 + 32 < K) issue(buf ^ 1, k0 + 32);  // overlap DMA with WMMA

    const __bf16* arow = A + (size_t)(mw + col) * K + k0;
    v16bf afr = frag_a(arow, half);
#pragma unroll
    for (int t = 0; t < 4; ++t) {
      v16bf bfr = frag_b(&Bs[buf][(t * 16 + col) * BS_LD], half);
      acc[t] = __builtin_amdgcn_wmma_f32_16x16x32_bf16(
          false, afr, false, bfr, (short)0, acc[t], false, false);
    }
  }

#pragma unroll
  for (int t = 0; t < 4; ++t) {
    int n = nb + t * 16 + col;
#pragma unroll
    for (int r = 0; r < 8; ++r)
      C[(size_t)(mw + r + 8 * half) * N + n] = acc[t][r];
  }
}

// ---------------------------------------------------------------------------
// RoPE + head split, f32 qkv -> bf16 Q(,pre-scaled 1/sqrt(64))/K/V [h][s][64]
// ---------------------------------------------------------------------------
__global__ __launch_bounds__(256) void rope_split(const float* __restrict__ qkv,
                                                  __bf16* __restrict__ Qh,
                                                  __bf16* __restrict__ Kh,
                                                  __bf16* __restrict__ Vh) {
  int tid = blockIdx.x * blockDim.x + threadIdx.x;
  if (tid >= SEQ * NHEAD * (HDIM / 2)) return;
  int s = tid >> 9, rem = tid & 511, h = rem >> 5, d2 = rem & 31;
  int d0 = 2 * d2, d1 = d0 + 1;

  const float* base = qkv + (size_t)s * (3 * EMB) + h * HDIM;
  float q0 = base[d0],           q1 = base[d1];
  float k0 = base[EMB + d0],     k1 = base[EMB + d1];
  float v0 = base[2 * EMB + d0], v1 = base[2 * EMB + d1];

  const float NLN = -0.28782313662425572f;  // -ln(10000)/32
  float fi0 = (float)((d0 < 32) ? d0 : d0 - 32);
  float fi1 = (float)((d1 < 32) ? d1 : d1 - 32);
  float a0 = (float)s * __expf(fi0 * NLN);
  float a1 = (float)s * __expf(fi1 * NLN);
  float c0 = __cosf(a0), s0 = __sinf(a0);
  float c1 = __cosf(a1), s1 = __sinf(a1);

  size_t o = (size_t)h * SEQ * HDIM + (size_t)s * HDIM;
  Qh[o + d0] = (__bf16)((q0 * c0 - q1 * s0) * 0.125f);
  Qh[o + d1] = (__bf16)((q1 * c1 + q0 * s1) * 0.125f);
  Kh[o + d0] = (__bf16)(k0 * c0 - k1 * s0);
  Kh[o + d1] = (__bf16)(k1 * c1 + k0 * s1);
  Vh[o + d0] = (__bf16)v0;
  Vh[o + d1] = (__bf16)v1;
}

// ---------------------------------------------------------------------------
// TDM: load a 32x64 bf16 tile (row stride 64 elems) into LDS with 16B row
// padding -> LDS stride 144B (72 elems). D# per ISA ch.8; clang-23 6-arg form.
// ---------------------------------------------------------------------------
#define KT_LD 72
__device__ __forceinline__ void tdm_load_tile(const __bf16* gsrc,
                                              unsigned lds_byte, int rows_rem) {
  unsigned long long ga = (unsigned long long)(uintptr_t)gsrc;
  v4u g0;
  g0[0] = 1u;                                   // count=1, user descriptor
  g0[1] = lds_byte;                             // lds_addr
  g0[2] = (unsigned)ga;                         // global_addr[31:0]
  g0[3] = (unsigned)((ga >> 32) & 0x1ffffffu)   // global_addr[56:32]
          | (2u << 30);                         // type=2 (image)
  v8i g1;
  g1[0] = (int)((1u << 16)    // data_size = 2B
                | (1u << 20)  // pad_enable
                | (4u << 22)  // pad_interval: 32 DWORDs
                | (3u << 25)); // pad_amount: 4 DWORDs (16B)
  g1[1] = (int)(64u << 16);                     // tensor_dim0 = 64 (bits 79:48)
  g1[2] = (int)(((unsigned)rows_rem & 0xffffu) << 16);  // tensor_dim1 lo
  g1[3] = (int)((((unsigned)rows_rem >> 16) & 0xffffu)  // tensor_dim1 hi
                | (64u << 16));                 // tile_dim0 = 64
  g1[4] = 32;                                   // tile_dim1=32, tile_dim2=0
  g1[5] = 64;                                   // tensor_dim0_stride = 64
  g1[6] = 0;
  g1[7] = 0;
  v4i z4 = {0, 0, 0, 0};
  v8i z8 = {0, 0, 0, 0, 0, 0, 0, 0};
  __builtin_amdgcn_tensor_load_to_lds(g0, g1, z4, z4, z8, 0);
}

// ---------------------------------------------------------------------------
// Flash attention, causal. grid=(head, qblock 64). 128 thr = 4 waves x 16 q.
// K tiles: TDM double-buffered; V tiles: cooperative transpose to LDS.
// ---------------------------------------------------------------------------
#define VT_LD 40
#define PB_LD 40

__global__ __launch_bounds__(128) void attn_fwd(const __bf16* __restrict__ Qh,
                                                const __bf16* __restrict__ Kh,
                                                const __bf16* __restrict__ Vh,
                                                __bf16* __restrict__ AO) {
  __shared__ __align__(16) __bf16 Kt[2][32 * KT_LD];
  __shared__ __align__(16) __bf16 Vt[2][64 * VT_LD];
  __shared__ __align__(16) __bf16 Pb[4][16 * PB_LD];

  const int h    = blockIdx.x;
  const int qb   = blockIdx.y * 64;
  const int tid  = threadIdx.x;
  const int lane = tid & 31;
  const int wave = tid >> 5;
  const int half = lane >> 4;
  const int col  = lane & 15;

  const __bf16* Qp = Qh + (size_t)h * SEQ * HDIM;
  const __bf16* Kp = Kh + (size_t)h * SEQ * HDIM;
  const __bf16* Vp = Vh + (size_t)h * SEQ * HDIM;
  const int qw = qb + wave * 16;
  const int q_last = qw + 15;

  // Q fragments (pre-scaled in rope kernel)
  const __bf16* qrow = Qp + (size_t)(qw + col) * HDIM;
  v16bf qf[2] = {frag_a(qrow, half), frag_a(qrow + 32, half)};

  // V tile: [j][d] global -> [d][j] LDS (transposed)
  auto fill_v = [&](int buf, int jb) {
    int j = tid >> 2, dq = (tid & 3) * 16;
    const __bf16* src = Vp + (size_t)(jb + j) * HDIM + dq;
#pragma unroll
    for (int i = 0; i < 16; ++i) Vt[buf][(dq + i) * VT_LD + j] = src[i];
  };

  v8f o[4] = {v8f{}, v8f{}, v8f{}, v8f{}};
  float mrow[8], lrow[8];
#pragma unroll
  for (int r = 0; r < 8; ++r) { mrow[r] = -__builtin_inff(); lrow[r] = 0.f; }

  const int ntiles = (qb + 64 + 31) / 32;
  if (wave == 0)
    tdm_load_tile(Kp, (unsigned)(uintptr_t)&Kt[0][0], SEQ);
  fill_v(0, 0);

  for (int jt = 0; jt < ntiles; ++jt) {
    const int jb = jt * 32, buf = jt & 1;
    if (wave == 0) __builtin_amdgcn_s_wait_tensorcnt(0);
    __syncthreads();
    if (jt + 1 < ntiles) {  // prefetch next tile while computing this one
      fill_v(buf ^ 1, jb + 32);
      if (wave == 0)
        tdm_load_tile(Kp + (size_t)(jb + 32) * HDIM,
                      (unsigned)(uintptr_t)&Kt[buf ^ 1][0], SEQ - (jb + 32));
    }

    if (jb <= q_last) {
      // scores: two 16-wide j subtiles, K=32 over d twice
      v8f sc[2] = {v8f{}, v8f{}};
#pragma unroll
      for (int sub = 0; sub < 2; ++sub) {
        const __bf16* krow = &Kt[buf][(sub * 16 + col) * KT_LD];
#pragma unroll
        for (int f = 0; f < 2; ++f) {
          v16bf kf = frag_b(krow + f * 32, half);
          sc[sub] = __builtin_amdgcn_wmma_f32_16x16x32_bf16(
              false, qf[f], false, kf, (short)0, sc[sub], false, false);
        }
      }
      // causal mask + online softmax
      const int k0i = jb + col, k1i = jb + 16 + col;
#pragma unroll
      for (int r = 0; r < 8; ++r) {
        int qi = qw + r + 8 * half;
        if (k0i > qi) sc[0][r] = -__builtin_inff();
        if (k1i > qi) sc[1][r] = -__builtin_inff();
      }
#pragma unroll
      for (int r = 0; r < 8; ++r) {
        float v = fmaxf(sc[0][r], sc[1][r]);
#pragma unroll
        for (int off = 1; off < 16; off <<= 1)
          v = fmaxf(v, __shfl_xor(v, off, 32));
        float mn = fmaxf(mrow[r], v);
        bool dead = (mn == -__builtin_inff());
        float alpha = dead ? 1.f : __expf(mrow[r] - mn);
        float p0 = dead ? 0.f : __expf(sc[0][r] - mn);
        float p1 = dead ? 0.f : __expf(sc[1][r] - mn);
        float rs = p0 + p1;
#pragma unroll
        for (int off = 1; off < 16; off <<= 1)
          rs += __shfl_xor(rs, off, 32);
        lrow[r] = lrow[r] * alpha + rs;
        mrow[r] = mn;
#pragma unroll
        for (int t = 0; t < 4; ++t) o[t][r] *= alpha;
        int m = r + 8 * half;
        Pb[wave][m * PB_LD + col]      = (__bf16)p0;
        Pb[wave][m * PB_LD + 16 + col] = (__bf16)p1;
      }
      // O += P(16x32) x V(32x64)
      const __bf16* prow = &Pb[wave][col * PB_LD];
      v16bf pa = frag2x16(prow + 8 * half, prow + 16 + 8 * half);
#pragma unroll
      for (int t = 0; t < 4; ++t) {
        v16bf vf = frag_b(&Vt[buf][(t * 16 + col) * VT_LD], half);
        o[t] = __builtin_amdgcn_wmma_f32_16x16x32_bf16(
            false, pa, false, vf, (short)0, o[t], false, false);
      }
    }
    __syncthreads();
  }

  // normalize, write bf16 [s][EMB] for the output projection
#pragma unroll
  for (int t = 0; t < 4; ++t) {
    int d = t * 16 + col;
#pragma unroll
    for (int r = 0; r < 8; ++r) {
      int m = qw + r + 8 * half;
      AO[(size_t)m * EMB + h * HDIM + d] = (__bf16)(o[t][r] / lrow[r]);
    }
  }
}

// ---------------------------------------------------------------------------
extern "C" void kernel_launch(void* const* d_in, const int* in_sizes, int n_in,
                              void* d_out, int out_size, void* d_ws, size_t ws_size,
                              hipStream_t stream) {
  const float* x    = (const float*)d_in[0];
  const float* Wqkv = (const float*)d_in[1];
  const float* Wout = (const float*)d_in[2];
  float* out = (float*)d_out;

  char* w = (char*)d_ws;
  __bf16* xb  = (__bf16*)w;                       w += (size_t)SEQ * EMB * 2;
  __bf16* wqb = (__bf16*)w;                       w += (size_t)3 * EMB * EMB * 2;
  __bf16* wob = (__bf16*)w;                       w += (size_t)EMB * EMB * 2;
  float*  qkv = (float*)w;                        w += (size_t)SEQ * 3 * EMB * 4;
  __bf16* Qhp = (__bf16*)w;                       w += (size_t)SEQ * EMB * 2;
  __bf16* Khp = (__bf16*)w;                       w += (size_t)SEQ * EMB * 2;
  __bf16* Vhp = (__bf16*)w;                       w += (size_t)SEQ * EMB * 2;
  __bf16* AOb = (__bf16*)w;

  cvt4<<<(SEQ * EMB / 4 + 255) / 256, 256, 0, stream>>>(x, xb, SEQ * EMB / 4);
  cvt4<<<(3 * EMB * EMB / 4 + 255) / 256, 256, 0, stream>>>(Wqkv, wqb, 3 * EMB * EMB / 4);
  cvt4<<<(EMB * EMB / 4 + 255) / 256, 256, 0, stream>>>(Wout, wob, EMB * EMB / 4);

  gemm_bf16<<<dim3((3 * EMB) / 64, SEQ / 128), 256, 0, stream>>>(
      xb, wqb, qkv, SEQ, 3 * EMB, EMB);
  rope_split<<<(SEQ * NHEAD * (HDIM / 2)) / 256, 256, 0, stream>>>(
      qkv, Qhp, Khp, Vhp);
  attn_fwd<<<dim3(NHEAD, SEQ / 64), 128, 0, stream>>>(Qhp, Khp, Vhp, AOb);
  gemm_bf16<<<dim3(EMB / 64, SEQ / 128), 256, 0, stream>>>(
      AOb, wob, out, SEQ, EMB, EMB);
}
